// EdgeModel_50371376447949
// MI455X (gfx1250) — matrix operand order repaired and checked by reading, not compiled, not run on hardware
//
#include <hip/hip_runtime.h>
#include <hip/hip_bf16.h>
#include <stdint.h>

typedef __bf16 bf16_t;
typedef __attribute__((ext_vector_type(16))) __bf16 v16bf;
typedef __attribute__((ext_vector_type(8)))  __bf16 v8bf;
typedef __attribute__((ext_vector_type(8)))  float  v8f;
typedef __attribute__((ext_vector_type(4)))  float  v4f;

#define BN_EPS 1e-5f

static __device__ __forceinline__ v8f wmma_bf16(v16bf a, v16bf b, v8f c) {
    return __builtin_amdgcn_wmma_f32_16x16x32_bf16(false, a, false, b,
                                                   (short)0, c, false, false);
}

// gfx1250 async global->LDS copy (VGLOBAL GLOBAL_LOAD_ASYNC_TO_LDS_B128, GV mode).
// ldsoff: per-lane LDS byte offset (low 32 bits of generic shared pointer).
static __device__ __forceinline__ void async_load_lds_b128(uint32_t ldsoff,
                                                           const void* gaddr) {
    asm volatile("global_load_async_to_lds_b128 %0, %1, off"
                 :: "v"(ldsoff), "v"(gaddr) : "memory");
}

static __device__ __forceinline__ void wait_asynccnt0() {
#if __has_builtin(__builtin_amdgcn_s_wait_asynccnt)
    __builtin_amdgcn_s_wait_asynccnt(0);
#else
    asm volatile("s_wait_asynccnt 0x0" ::: "memory");
#endif
}

// ---------------------------------------------------------------------------
// Pack W (K x 256, row-major f32) into bf16 B-fragment layout:
// fragment (kt, nt): lane l holds W[k, n] with n = nt*16 + (l & 15),
// k = kt*32 + 8*(l >> 4) + j for j < 8, kt*32 + 16 + 8*(l >> 4) + (j - 8) for j >= 8.
// packed index = ((kt*16 + nt)*32 + l)*16 + j
// One K-step (16 fragments) is a contiguous 16 KB block -> trivial LDS stage.
// ---------------------------------------------------------------------------
__global__ void pack_w_kernel(const float* __restrict__ W, bf16_t* __restrict__ Wp) {
    int k = blockIdx.x;
    int n = threadIdx.x;
    int kt  = k >> 5;
    int kin = k & 31;
    int l   = ((kin & 8) << 1) + (n & 15);
    int j   = (kin & 7) + ((kin & 16) >> 1);
    int nt  = n >> 4;
    Wp[(((size_t)(kt * 16 + nt) * 32) + l) * 16 + j] = (bf16_t)W[(size_t)k * 256 + n];
}

__global__ void zero_stats_kernel(float* __restrict__ s) {
    s[blockIdx.x * blockDim.x + threadIdx.x] = 0.0f;
}

// ---------------------------------------------------------------------------
// GEMM1: virtual-concat [E x 384] @ W1 -> Y1 (bf16, C-fragment layout) + stats.
// Block = 256 threads (8 waves) = 8 row-tiles x 256 cols.
// Wave = 2 row-tiles (32 rows) x 8 n-tiles (128 cols).
// B staged per-K-step in double-buffered LDS via async global->LDS DMA.
// ---------------------------------------------------------------------------
__global__ __launch_bounds__(256, 1) void gemm1_kernel(
    const float* __restrict__ src, const float* __restrict__ dst,
    const float* __restrict__ edge, const float* __restrict__ u,
    const int*  __restrict__ batch,
    const bf16_t* __restrict__ Wp1,
    bf16_t* __restrict__ y1,
    float* __restrict__ sum1, float* __restrict__ ss1,
    int nRowTiles)
{
    __shared__ __align__(32) bf16_t bstage[2][16 * 512];  // 2 x 16 KB
    __shared__ float lsum[256];
    __shared__ float lss[256];

    const int tid   = threadIdx.x;
    const int lane  = tid & 31;
    const int w     = tid >> 5;
    const int nhalf = w & 1;                  // which 128 columns
    lsum[tid] = 0.0f; lss[tid] = 0.0f;

    const int rt0  = blockIdx.x * 8 + (w >> 1) * 2;
    const int rt1  = rt0 + 1;
    const bool act0 = rt0 < nRowTiles;
    const bool act1 = rt1 < nRowTiles;
    const int rc0 = act0 ? rt0 : 0;
    const int rc1 = act1 ? rt1 : 0;

    const int mrow = lane & 15;
    const int sel  = (lane & 16) >> 1;        // 0 or 8
    const int m0 = rc0 * 16 + mrow;
    const int m1 = rc1 * 16 + mrow;

    const float* psrc0 = src  + (size_t)m0 * 128;
    const float* pdst0 = dst  + (size_t)m0 * 128;
    const float* pedg0 = edge + (size_t)m0 * 64;
    const float* pu0   = u    + (size_t)batch[m0] * 64;
    const float* psrc1 = src  + (size_t)m1 * 128;
    const float* pdst1 = dst  + (size_t)m1 * 128;
    const float* pedg1 = edge + (size_t)m1 * 64;
    const float* pu1   = u    + (size_t)batch[m1] * 64;

    const uint32_t ldsbase = (uint32_t)(uintptr_t)(&bstage[0][0]);

    auto stage = [&](int buf, int kt) {        // async DMA 16 KB, no VGPR traffic
        const char* g = (const char*)(Wp1 + (size_t)kt * 8192) + (size_t)tid * 16;
        uint32_t l0 = ldsbase + (uint32_t)buf * 16384u + (uint32_t)tid * 16u;
        #pragma unroll
        for (int i = 0; i < 4; ++i)
            async_load_lds_b128(l0 + (uint32_t)i * 4096u, g + (size_t)i * 4096);
    };

    auto build_a = [&](const float* ps, const float* pd, const float* pe,
                       const float* pu, int kt) -> v16bf {
        v16bf a;
        #pragma unroll
        for (int ch = 0; ch < 2; ++ch) {
            const int cbase = kt * 32 + ch * 16;   // compile-time after unroll
            const float* p; int off;
            if      (cbase < 128) { p = ps; off = cbase;       }
            else if (cbase < 256) { p = pd; off = cbase - 128; }
            else if (cbase < 320) { p = pe; off = cbase - 256; }
            else                  { p = pu; off = cbase - 320; }
            const float* q = p + off + sel;        // 32B aligned
            v4f f0 = *(const v4f*)q;
            v4f f1 = *(const v4f*)(q + 4);
            #pragma unroll
            for (int j = 0; j < 4; ++j) {
                a[ch * 8 + j]     = (bf16_t)f0[j];
                a[ch * 8 + 4 + j] = (bf16_t)f1[j];
            }
        }
        return a;
    };

    v8f acc0[8], acc1[8];
    #pragma unroll
    for (int i = 0; i < 8; ++i) {
        acc0[i] = v8f{0.f,0.f,0.f,0.f,0.f,0.f,0.f,0.f};
        acc1[i] = v8f{0.f,0.f,0.f,0.f,0.f,0.f,0.f,0.f};
    }

    stage(0, 0);
    wait_asynccnt0();
    __syncthreads();

    #pragma unroll
    for (int kt = 0; kt < 12; ++kt) {
        const int buf = kt & 1;
        if (kt + 1 < 12) stage(buf ^ 1, kt + 1);   // DMA hidden behind 16 WMMAs

        v16bf a0 = build_a(psrc0, pdst0, pedg0, pu0, kt);
        v16bf a1 = build_a(psrc1, pdst1, pedg1, pu1, kt);

        const bf16_t* bb = bstage[buf] + (size_t)nhalf * 8 * 512;
        v16bf bcur = *(const v16bf*)(bb + lane * 16);
        #pragma unroll
        for (int nt = 0; nt < 8; ++nt) {
            v16bf bnext = (nt < 7) ? *(const v16bf*)(bb + (nt + 1) * 512 + lane * 16)
                                   : bcur;
            acc0[nt] = wmma_bf16(a0, bcur, acc0[nt]);
            acc1[nt] = wmma_bf16(a1, bcur, acc1[nt]);
            bcur = bnext;
        }
        wait_asynccnt0();  // this wave's DMA for next K-step has landed
        __syncthreads();   // block-wide: reads of buf done, next stage may overwrite
    }

    // Y1 in C-fragment layout: index ((rt*16 + gnt)*8 + v)*32 + lane.
    auto store_tile = [&](int rt, v8f* acc) {
        #pragma unroll
        for (int nt = 0; nt < 8; ++nt) {
            const int gnt = nhalf * 8 + nt;
            float s = 0.0f, s2 = 0.0f;
            #pragma unroll
            for (int v = 0; v < 8; ++v) {
                float c = acc[nt][v];
                s  += c;
                s2 += c * c;
                y1[((size_t)(rt * 16 + gnt) * 8 + v) * 32 + lane] = (bf16_t)c;
            }
            const int col = gnt * 16 + (lane & 15);
            atomicAdd(&lsum[col], s);
            atomicAdd(&lss[col],  s2);
        }
    };
    if (act0) store_tile(rt0, acc0);
    if (act1) store_tile(rt1, acc1);

    __syncthreads();
    atomicAdd(&sum1[tid], lsum[tid]);
    atomicAdd(&ss1[tid],  lss[tid]);
}

// mean/var -> fused affine: y' = y*scale + bias
__global__ void finalize_bn_kernel(const float* __restrict__ sum, const float* __restrict__ ss,
                                   const float* __restrict__ g, const float* __restrict__ b,
                                   float* __restrict__ scale, float* __restrict__ bias,
                                   float invE)
{
    int n = threadIdx.x;
    float mean = sum[n] * invE;
    float var  = ss[n] * invE - mean * mean;
    float r    = rsqrtf(var + BN_EPS);
    float sc   = r * g[n];
    scale[n] = sc;
    bias[n]  = b[n] - mean * sc;
}

// ---------------------------------------------------------------------------
// GEMM2: relu(bn1(Y1)) @ W2 -> d_out (pre-BN2) + stats. Same blocking.
// ---------------------------------------------------------------------------
__global__ __launch_bounds__(256, 1) void gemm2_kernel(
    const bf16_t* __restrict__ y1, const bf16_t* __restrict__ Wp2,
    const float* __restrict__ scaleA, const float* __restrict__ biasA,
    float* __restrict__ out,
    float* __restrict__ sum2, float* __restrict__ ss2,
    int nRowTiles)
{
    __shared__ __align__(32) bf16_t bstage[2][16 * 512];
    __shared__ float lsum[256];
    __shared__ float lss[256];

    const int tid   = threadIdx.x;
    const int lane  = tid & 31;
    const int w     = tid >> 5;
    const int nhalf = w & 1;
    lsum[tid] = 0.0f; lss[tid] = 0.0f;

    const int rt0  = blockIdx.x * 8 + (w >> 1) * 2;
    const int rt1  = rt0 + 1;
    const bool act0 = rt0 < nRowTiles;
    const bool act1 = rt1 < nRowTiles;
    const int rc0 = act0 ? rt0 : 0;
    const int rc1 = act1 ? rt1 : 0;

    const int mloc = lane & 15;
    const int sel  = (lane & 16) >> 1;         // 0 or 8
    const int vrow = mloc & 7;
    const int slhi = (mloc >> 3) << 4;         // 0 or 16

    // Y1 C-fragment layout: ((rt*16 + q)*8 + vrow)*32 + sel + slhi, q = kt*2+ch
    const bf16_t* a0base = y1 + ((size_t)(rc0 * 16) * 8 + vrow) * 32 + sel + slhi;
    const bf16_t* a1base = y1 + ((size_t)(rc1 * 16) * 8 + vrow) * 32 + sel + slhi;

    const uint32_t ldsbase = (uint32_t)(uintptr_t)(&bstage[0][0]);

    auto stage = [&](int buf, int kt) {
        const char* g = (const char*)(Wp2 + (size_t)kt * 8192) + (size_t)tid * 16;
        uint32_t l0 = ldsbase + (uint32_t)buf * 16384u + (uint32_t)tid * 16u;
        #pragma unroll
        for (int i = 0; i < 4; ++i)
            async_load_lds_b128(l0 + (uint32_t)i * 4096u, g + (size_t)i * 4096);
    };

    auto build_a = [&](const bf16_t* abase, int kt) -> v16bf {
        v16bf a;
        #pragma unroll
        for (int ch = 0; ch < 2; ++ch) {
            const int c = kt * 32 + ch * 16 + sel;       // first feature index
            v8bf raw = *(const v8bf*)(abase + (size_t)(kt * 2 + ch) * 256);
            #pragma unroll
            for (int j = 0; j < 8; ++j) {
                float f = (float)raw[j];
                f = f * scaleA[c + j] + biasA[c + j];    // BN1 affine
                f = f > 0.0f ? f : 0.0f;                 // ReLU
                a[ch * 8 + j] = (bf16_t)f;
            }
        }
        return a;
    };

    v8f acc0[8], acc1[8];
    #pragma unroll
    for (int i = 0; i < 8; ++i) {
        acc0[i] = v8f{0.f,0.f,0.f,0.f,0.f,0.f,0.f,0.f};
        acc1[i] = v8f{0.f,0.f,0.f,0.f,0.f,0.f,0.f,0.f};
    }

    stage(0, 0);
    wait_asynccnt0();
    __syncthreads();

    #pragma unroll
    for (int kt = 0; kt < 8; ++kt) {
        const int buf = kt & 1;
        if (kt + 1 < 8) stage(buf ^ 1, kt + 1);

        v16bf a0 = build_a(a0base, kt);
        v16bf a1 = build_a(a1base, kt);

        const bf16_t* bb = bstage[buf] + (size_t)nhalf * 8 * 512;
        v16bf bcur = *(const v16bf*)(bb + lane * 16);
        #pragma unroll
        for (int nt = 0; nt < 8; ++nt) {
            v16bf bnext = (nt < 7) ? *(const v16bf*)(bb + (nt + 1) * 512 + lane * 16)
                                   : bcur;
            acc0[nt] = wmma_bf16(a0, bcur, acc0[nt]);
            acc1[nt] = wmma_bf16(a1, bcur, acc1[nt]);
            bcur = bnext;
        }
        wait_asynccnt0();
        __syncthreads();
    }

    auto store_tile = [&](int rt, v8f* acc) {
        float* orow = out + ((size_t)rt * 16 + ((lane & 16) >> 1)) * 256 + (lane & 15);
        #pragma unroll
        for (int nt = 0; nt < 8; ++nt) {
            const int gnt = nhalf * 8 + nt;
            float s = 0.0f, s2 = 0.0f;
            #pragma unroll
            for (int v = 0; v < 8; ++v) {
                float c = acc[nt][v];
                s  += c;
                s2 += c * c;
                orow[(size_t)v * 256 + gnt * 16] = c;    // row-major pre-BN2
            }
            const int col = gnt * 16 + (lane & 15);
            atomicAdd(&lsum[col], s);
            atomicAdd(&lss[col],  s2);
        }
    };
    if (act0) store_tile(rt0, acc0);
    if (act1) store_tile(rt1, acc1);

    __syncthreads();
    atomicAdd(&sum2[tid], lsum[tid]);
    atomicAdd(&ss2[tid],  lss[tid]);
}

// In-place BN2 on d_out, vectorized 4 floats/thread.
__global__ void bn2_apply_kernel(float* __restrict__ out,
                                 const float* __restrict__ scale2,
                                 const float* __restrict__ bias2,
                                 size_t n4)
{
    size_t i = (size_t)blockIdx.x * blockDim.x + threadIdx.x;
    if (i >= n4) return;
    v4f x = ((v4f*)out)[i];
    int n0 = (int)((i & 63) << 2);
    #pragma unroll
    for (int j = 0; j < 4; ++j) x[j] = x[j] * scale2[n0 + j] + bias2[n0 + j];
    ((v4f*)out)[i] = x;
}

// ---------------------------------------------------------------------------
extern "C" void kernel_launch(void* const* d_in, const int* in_sizes, int n_in,
                              void* d_out, int out_size, void* d_ws, size_t ws_size,
                              hipStream_t stream) {
    const float* src   = (const float*)d_in[0];
    const float* dst   = (const float*)d_in[1];
    const float* edge  = (const float*)d_in[2];
    const float* u     = (const float*)d_in[3];
    const int*   batch = (const int*)d_in[4];
    const float* W1    = (const float*)d_in[5];
    const float* g1    = (const float*)d_in[6];
    const float* b1    = (const float*)d_in[7];
    const float* W2    = (const float*)d_in[8];
    const float* g2    = (const float*)d_in[9];
    const float* b2    = (const float*)d_in[10];
    float* out = (float*)d_out;

    const int E        = in_sizes[0] / 128;   // 500000
    const int rowTiles = E / 16;              // 31250 (E % 16 == 0)

    // Workspace layout
    char*   ws     = (char*)d_ws;
    bf16_t* Wp1    = (bf16_t*)ws;                       // 384*256 bf16 = 196608 B
    bf16_t* Wp2    = (bf16_t*)(ws + 196608);            // 256*256 bf16 = 131072 B
    float*  stats  = (float*)(ws + 327680);             // 2048 f32
    float*  sum1   = stats;         float* ss1   = stats + 256;
    float*  sum2   = stats + 512;   float* ss2   = stats + 768;
    float*  scaleA = stats + 1024;  float* biasA = stats + 1280;
    float*  scale2 = stats + 1536;  float* bias2 = stats + 1792;
    bf16_t* y1     = (bf16_t*)(ws + 335872);            // E*256 bf16 (~256 MB)

    pack_w_kernel<<<384, 256, 0, stream>>>(W1, Wp1);
    pack_w_kernel<<<256, 256, 0, stream>>>(W2, Wp2);
    zero_stats_kernel<<<4, 256, 0, stream>>>(stats);

    const int nBlocks = (rowTiles + 7) / 8;
    gemm1_kernel<<<nBlocks, 256, 0, stream>>>(src, dst, edge, u, batch, Wp1,
                                              y1, sum1, ss1, rowTiles);
    finalize_bn_kernel<<<1, 256, 0, stream>>>(sum1, ss1, g1, b1, scaleA, biasA,
                                              1.0f / (float)E);
    gemm2_kernel<<<nBlocks, 256, 0, stream>>>(y1, Wp2, scaleA, biasA, out,
                                              sum2, ss2, rowTiles);
    finalize_bn_kernel<<<1, 256, 0, stream>>>(sum2, ss2, g2, b2, scale2, bias2,
                                              1.0f / (float)E);

    size_t n4 = (size_t)E * 64;
    int bnBlocks = (int)((n4 + 255) / 256);
    bn2_apply_kernel<<<bnBlocks, 256, 0, stream>>>(out, scale2, bias2, n4);
}
